// Transformer_17274358464791
// MI455X (gfx1250) — matrix operand order repaired and checked by reading, not compiled, not run on hardware
//
#include <hip/hip_runtime.h>
#include <math.h>

#define NN 50000
#define EE 800000
#define HC_ 128
#define HEADS_ 4
#define EMB_ 32

typedef __attribute__((ext_vector_type(16))) _Float16 v16h;
typedef __attribute__((ext_vector_type(8)))  float    v8f;

// ---------------------------------------------------------------------------
// Generic fill
// ---------------------------------------------------------------------------
__global__ void fill_kernel(float* p, float val, int n) {
    int t = blockIdx.x * blockDim.x + threadIdx.x;
    if (t < n) p[t] = val;
}

// ---------------------------------------------------------------------------
// WMMA linear: Y[n][m] = act( sum_k X[n][k] * W[m][k] + b[m] )
// One wave computes a 16x16 tile of Y.  K stepped by 32 (K % 32 == 0).
// blockDim = 64 (2 waves).  grid = (Nrows/16, M/32).
// Fragment packing per CDNA5 ISA 7.12.2 (wave32):
//   A (16x32 f16): lane L: row = L%16, half = L/16;
//       a[j]   -> K = half*8 + j        (j=0..7)
//       a[8+j] -> K = 16 + half*8 + j
//   B (32x16 f16): lane L: col N = L%16, half = L/16; b[j] -> same K as a[j]
//   D (16x16 f32): lane L: c[i] -> row = i + 8*half, col = L%16
// ---------------------------------------------------------------------------
__global__ void wmma_linear_kernel(const float* __restrict__ X,
                                   const float* __restrict__ W,
                                   const float* __restrict__ bias,
                                   float* __restrict__ Y,
                                   int K, int M, int relu) {
    int lane = threadIdx.x & 31;
    int wv   = threadIdx.x >> 5;              // 0..1
    int row0 = blockIdx.x * 16;
    int m0   = (blockIdx.y * 2 + wv) * 16;
    int r    = lane & 15;
    int half = lane >> 4;

    const float* xrow = X + (size_t)(row0 + r) * K;
    const float* wrow = W + (size_t)(m0 + r) * K;

    v8f acc = {};
    for (int k0 = 0; k0 < K; k0 += 32) {
        v16h a, b;
#pragma unroll
        for (int j = 0; j < 8; ++j) {
            a[j]     = (_Float16)xrow[k0 + half * 8 + j];
            a[8 + j] = (_Float16)xrow[k0 + 16 + half * 8 + j];
        }
#pragma unroll
        for (int j = 0; j < 16; ++j)
            b[j] = (_Float16)wrow[k0 + (j < 8 ? half * 8 + j : 16 + half * 8 + (j - 8))];
        acc = __builtin_amdgcn_wmma_f32_16x16x32_f16(
            false, a, false, b, (short)0, acc, false, false);
    }

    float bv = bias ? bias[m0 + r] : 0.0f;
#pragma unroll
    for (int i = 0; i < 8; ++i) {
        int rr  = row0 + i + 8 * half;
        float o = acc[i] + bv;
        if (relu) o = fmaxf(o, 0.0f);
        Y[(size_t)rr * M + m0 + r] = o;
    }
}

// ---------------------------------------------------------------------------
// Float atomic max via int/uint ordering trick (matches global_atomic_{max,min})
// ---------------------------------------------------------------------------
__device__ inline void atomicMaxFloat(float* addr, float val) {
    if (val >= 0.0f)
        atomicMax((int*)addr, __float_as_int(val));
    else
        atomicMin((unsigned int*)addr, __float_as_uint(val));
}

__device__ inline float waveReduceSum(float p) {
#pragma unroll
    for (int off = 16; off; off >>= 1) p += __shfl_xor(p, off, 32);
    return p;
}

// ---------------------------------------------------------------------------
// Pass 1: per-edge attention logits + segment max.
// One wave32 per edge; lane i handles channel h*32+i for each head h.
// We (HCxEDGE_DIM=128x16) cached transposed in LDS (conflict-free).
// ---------------------------------------------------------------------------
__global__ void edge_alpha_kernel(const float* __restrict__ q,
                                  const float* __restrict__ k,
                                  const float* __restrict__ edge_attr,
                                  const float* __restrict__ We,
                                  const int* __restrict__ ei,
                                  float* __restrict__ alpha,
                                  float* __restrict__ amax) {
    __shared__ float sWe[16 * HC_];           // sWe[d*128 + j] = We[j][d]
    for (int t = threadIdx.x; t < 16 * HC_; t += blockDim.x) {
        int j = t >> 4, d = t & 15;
        sWe[d * HC_ + j] = We[t];
    }
    __syncthreads();

    int lane = threadIdx.x & 31;
    int e = blockIdx.x * (blockDim.x >> 5) + (threadIdx.x >> 5);
    if (e >= EE) return;
    int src = ei[e], dst = ei[EE + e];

    float ea[16];
#pragma unroll
    for (int d = 0; d < 16; ++d) ea[d] = edge_attr[(size_t)e * 16 + d];

    const float scale = 0.17677669529663687f; // 1/sqrt(32)
#pragma unroll
    for (int h = 0; h < HEADS_; ++h) {
        int idx = h * 32 + lane;
        float ev = 0.0f;
#pragma unroll
        for (int d = 0; d < 16; ++d) ev += ea[d] * sWe[d * HC_ + idx];
        float p = q[(size_t)dst * HC_ + idx] * (k[(size_t)src * HC_ + idx] + ev);
        p = waveReduceSum(p);
        if (lane == 0) {
            float al = p * scale;
            alpha[(size_t)e * HEADS_ + h] = al;
            atomicMaxFloat(&amax[(size_t)dst * HEADS_ + h], al);
        }
    }
}

// ---------------------------------------------------------------------------
// Pass 2: alpha <- exp(alpha - amax[dst]); denom[dst] += alpha.  Thread per (e,h).
// ---------------------------------------------------------------------------
__global__ void edge_softmax_kernel(const int* __restrict__ ei,
                                    float* __restrict__ alpha,
                                    const float* __restrict__ amax,
                                    float* __restrict__ denom) {
    int t = blockIdx.x * blockDim.x + threadIdx.x;
    if (t >= EE * HEADS_) return;
    int e = t >> 2, h = t & 3;
    int dst = ei[EE + e];
    float a = __expf(alpha[t] - amax[(size_t)dst * HEADS_ + h]);
    alpha[t] = a;
    atomicAdd(&denom[(size_t)dst * HEADS_ + h], a);
}

// ---------------------------------------------------------------------------
// Pass 3: out[dst] += (v[src] + e) * alpha/denom[dst].  One wave32 per edge.
// ---------------------------------------------------------------------------
__global__ void edge_scatter_kernel(const float* __restrict__ v,
                                    const float* __restrict__ edge_attr,
                                    const float* __restrict__ We,
                                    const int* __restrict__ ei,
                                    const float* __restrict__ alpha,
                                    const float* __restrict__ denom,
                                    float* __restrict__ outm) {
    __shared__ float sWe[16 * HC_];
    for (int t = threadIdx.x; t < 16 * HC_; t += blockDim.x) {
        int j = t >> 4, d = t & 15;
        sWe[d * HC_ + j] = We[t];
    }
    __syncthreads();

    int lane = threadIdx.x & 31;
    int e = blockIdx.x * (blockDim.x >> 5) + (threadIdx.x >> 5);
    if (e >= EE) return;
    int src = ei[e], dst = ei[EE + e];

    float ea[16];
#pragma unroll
    for (int d = 0; d < 16; ++d) ea[d] = edge_attr[(size_t)e * 16 + d];

#pragma unroll
    for (int h = 0; h < HEADS_; ++h) {
        int idx = h * 32 + lane;
        float ev = 0.0f;
#pragma unroll
        for (int d = 0; d < 16; ++d) ev += ea[d] * sWe[d * HC_ + idx];
        float a = alpha[(size_t)e * HEADS_ + h] /
                  (denom[(size_t)dst * HEADS_ + h] + 1e-16f);
        float m = (v[(size_t)src * HC_ + idx] + ev) * a;
        atomicAdd(&outm[(size_t)dst * HC_ + idx], m);
    }
}

// ---------------------------------------------------------------------------
// Beta gate: b = sigmoid([out, skip, out-skip] . Wb); h = b*skip + (1-b)*out
// One wave32 per node.
// ---------------------------------------------------------------------------
__global__ void beta_kernel(const float* __restrict__ outm,
                            const float* __restrict__ skip,
                            const float* __restrict__ Wb,
                            float* __restrict__ hconv) {
    int lane = threadIdx.x & 31;
    int n = blockIdx.x * (blockDim.x >> 5) + (threadIdx.x >> 5);
    if (n >= NN) return;

    float o[HEADS_], s[HEADS_];
    float p = 0.0f;
#pragma unroll
    for (int h = 0; h < HEADS_; ++h) {
        int idx = h * 32 + lane;
        o[h] = outm[(size_t)n * HC_ + idx];
        s[h] = skip[(size_t)n * HC_ + idx];
        p += Wb[idx] * o[h] + Wb[HC_ + idx] * s[h] + Wb[2 * HC_ + idx] * (o[h] - s[h]);
    }
    p = waveReduceSum(p);
    float b = 1.0f / (1.0f + __expf(-p));
#pragma unroll
    for (int h = 0; h < HEADS_; ++h) {
        int idx = h * 32 + lane;
        hconv[(size_t)n * HC_ + idx] = b * s[h] + (1.0f - b) * o[h];
    }
}

// ---------------------------------------------------------------------------
// Pooling
// ---------------------------------------------------------------------------
__global__ void pool_accum_kernel(const float* __restrict__ h,
                                  const int* __restrict__ batch,
                                  float* __restrict__ sums,
                                  float* __restrict__ cnt) {
    int t = blockIdx.x * blockDim.x + threadIdx.x;
    if (t >= NN * EMB_) return;
    int n = t >> 5, c = t & 31;
    int g = batch[n];
    atomicAdd(&sums[g * EMB_ + c], h[(size_t)n * EMB_ + c]);
    if (c == 0) atomicAdd(&cnt[g], 1.0f);
}

__global__ void pool_out_kernel(const float* __restrict__ sums,
                                const float* __restrict__ cnt,
                                const float* __restrict__ Wout,
                                const float* __restrict__ bout,
                                float* __restrict__ out) {
    int g = blockIdx.x;
    int lane = threadIdx.x;
    float p = (sums[g * EMB_ + lane] / fmaxf(cnt[g], 1.0f)) * Wout[lane];
    p = waveReduceSum(p);
    if (lane == 0) out[g] = p + bout[0];
}

// ---------------------------------------------------------------------------
// Host launch
// ---------------------------------------------------------------------------
static inline int cdiv(int a, int b) { return (a + b - 1) / b; }

extern "C" void kernel_launch(void* const* d_in, const int* in_sizes, int n_in,
                              void* d_out, int out_size, void* d_ws, size_t ws_size,
                              hipStream_t stream) {
    (void)in_sizes; (void)n_in; (void)out_size; (void)ws_size;
    const int N = NN, E = EE;

    const float* x         = (const float*)d_in[0];
    const float* edge_attr = (const float*)d_in[1];
    const int*   ei        = (const int*)d_in[2];
    const int*   batch     = (const int*)d_in[3];
    auto P = [&](int i) { return (const float*)d_in[i]; };
    // Leaf order (setup_inputs insertion order):
    // per layer: qW qb kW kb vW vb eW skipW skipb betaW tW tb  (12 leaves)

    float* ws    = (float*)d_ws;
    float* q     = ws;                       // N*128
    float* k     = q     + (size_t)N * HC_;  // N*128
    float* v     = k     + (size_t)N * HC_;  // N*128
    float* skip  = v     + (size_t)N * HC_;  // N*128
    float* outm  = skip  + (size_t)N * HC_;  // N*128
    float* hconv = outm  + (size_t)N * HC_;  // N*128
    float* hnext = hconv + (size_t)N * HC_;  // N*32
    float* alpha = hnext + (size_t)N * EMB_; // E*4
    float* amax  = alpha + (size_t)E * HEADS_; // N*4
    float* denom = amax  + (size_t)N * HEADS_; // N*4
    float* sums  = denom + (size_t)N * HEADS_; // 64*32
    float* cnt   = sums  + 64 * EMB_;          // 64

    for (int layer = 0; layer < 2; ++layer) {
        int base = 4 + layer * 12;
        const float* X = (layer == 0) ? x : hnext;
        int K = (layer == 0) ? 128 : 32;

        dim3 gp(N / 16, HC_ / 32);
        wmma_linear_kernel<<<gp, 64, 0, stream>>>(X, P(base + 0), P(base + 1), q,    K, HC_, 0);
        wmma_linear_kernel<<<gp, 64, 0, stream>>>(X, P(base + 2), P(base + 3), k,    K, HC_, 0);
        wmma_linear_kernel<<<gp, 64, 0, stream>>>(X, P(base + 4), P(base + 5), v,    K, HC_, 0);
        wmma_linear_kernel<<<gp, 64, 0, stream>>>(X, P(base + 7), P(base + 8), skip, K, HC_, 0);

        fill_kernel<<<cdiv(N * HEADS_, 256), 256, 0, stream>>>(amax, -INFINITY, N * HEADS_);
        fill_kernel<<<cdiv(N * HEADS_, 256), 256, 0, stream>>>(denom, 0.0f, N * HEADS_);
        fill_kernel<<<cdiv(N * HC_, 256), 256, 0, stream>>>(outm, 0.0f, N * HC_);

        edge_alpha_kernel<<<cdiv(E, 8), 256, 0, stream>>>(q, k, edge_attr, P(base + 6),
                                                          ei, alpha, amax);
        edge_softmax_kernel<<<cdiv(E * HEADS_, 256), 256, 0, stream>>>(ei, alpha, amax, denom);
        edge_scatter_kernel<<<cdiv(E, 8), 256, 0, stream>>>(v, edge_attr, P(base + 6),
                                                            ei, alpha, denom, outm);
        beta_kernel<<<cdiv(N, 8), 256, 0, stream>>>(outm, skip, P(base + 9), hconv);

        // transf: [N,128] -> relu([N,32])
        wmma_linear_kernel<<<dim3(N / 16, EMB_ / 32), 64, 0, stream>>>(
            hconv, P(base + 10), P(base + 11), hnext, HC_, EMB_, 1);
    }

    fill_kernel<<<cdiv(64 * EMB_ + 64, 256), 256, 0, stream>>>(sums, 0.0f, 64 * EMB_ + 64);
    pool_accum_kernel<<<cdiv(N * EMB_, 256), 256, 0, stream>>>(hnext, batch, sums, cnt);
    pool_out_kernel<<<64, 32, 0, stream>>>(sums, cnt, P(28), P(29), (float*)d_out);
}
